// VectorQuantizer_21938692948238
// MI455X (gfx1250) — matrix operand order repaired and checked by reading, not compiled, not run on hardware
//
#include <hip/hip_runtime.h>
#include <stdint.h>

typedef __bf16 bf16;
typedef __attribute__((ext_vector_type(16))) __bf16 v16bf;
typedef __attribute__((ext_vector_type(8)))  float  v8f;
typedef __attribute__((ext_vector_type(4)))  unsigned int u32x4;
typedef __attribute__((ext_vector_type(4)))  int i32x4;
typedef __attribute__((ext_vector_type(8)))  int i32x8;

#define N_ROWS 65536
#define DIM 64
#define K_CODES 1024
#define ROWS_PER_WAVE 32
#define ROWS_PER_BLOCK 256
#define THREADS 256
#define MAIN_BLOCKS (N_ROWS / ROWS_PER_BLOCK) /* 256 */

union V16U { v16bf v; uint4 q[2]; };

#define WMMA_BF16(A, B, C) \
  __builtin_amdgcn_wmma_f32_16x16x32_bf16(false, (A), false, (B), (short)0, (C), false, false)

// ---- prep: split codebook f32 -> bf16 hi + bf16 lo (stored as u16 bits) ----
__global__ __launch_bounds__(256)
void vq_prep_split(const float* __restrict__ cb,
                   unsigned short* __restrict__ hi,
                   unsigned short* __restrict__ lo) {
  int i = blockIdx.x * 256 + threadIdx.x;
  if (i < K_CODES * DIM) {
    float v = cb[i];
    bf16 h = (bf16)v;
    bf16 l = (bf16)(v - (float)h);
    hi[i] = __builtin_bit_cast(unsigned short, h);
    lo[i] = __builtin_bit_cast(unsigned short, l);
  }
}

// ---- prep: hn[k] = -0.5 * ||c_k||^2 (exact f32) ----
__global__ __launch_bounds__(256)
void vq_prep_norm(const float* __restrict__ cb, float* __restrict__ hn) {
  int k = blockIdx.x * 256 + threadIdx.x;
  if (k < K_CODES) {
    const float4* p = (const float4*)(cb + k * DIM);
    float s = 0.f;
#pragma unroll
    for (int j = 0; j < DIM / 4; ++j) {
      float4 f = p[j];
      s += f.x * f.x + f.y * f.y + f.z * f.z + f.w * f.w;
    }
    hn[k] = -0.5f * s;
  }
}

// TDM: async 1D copy of the 128KB bf16-hi codebook, global -> LDS.
// D# per CDNA5 ISA ch.8: group0 {count, lds_addr, global_addr[56:0], type=2},
// group1 {data_size=8B, tensor_dim0=tile_dim0=16384, dim1=1, stride0=16384}.
__device__ __forceinline__ void tdm_stage_cbhi(const unsigned short* src, unsigned lds_off) {
  unsigned long long ga = (unsigned long long)(uintptr_t)src;
  u32x4 g0;
  g0[0] = 1u;                                               // count=1, user mode
  g0[1] = lds_off;                                          // lds_addr (bytes)
  g0[2] = (unsigned)(ga & 0xFFFFFFFFu);                     // global_addr[31:0]
  g0[3] = (unsigned)((ga >> 32) & 0x1FFFFFFu) | (2u << 30); // global_addr[56:32], type=2
  i32x8 g1;
  g1[0] = (int)(3u << 16);        // workgroup_mask=0, data_size=8B, no barrier/iter/pad
  g1[1] = (int)(16384u << 16);    // tensor_dim0[15:0]=16384 (8B elems)
  g1[2] = (int)(1u << 16);        // tensor_dim0[31:16]=0, tensor_dim1[15:0]=1
  g1[3] = (int)(16384u << 16);    // tensor_dim1[31:16]=0, tile_dim0=16384
  g1[4] = 1;                      // tile_dim1=1, tile_dim2=0
  g1[5] = 16384;                  // tensor_dim0_stride[31:0]
  g1[6] = 0;                      // stride0 hi, stride1 lo
  g1[7] = 0;
  i32x4 z4 = {0, 0, 0, 0};
#if __clang_major__ >= 23
  i32x8 z8 = {0, 0, 0, 0, 0, 0, 0, 0};
  __builtin_amdgcn_tensor_load_to_lds(g0, g1, z4, z4, z8, 0);
#else
  __builtin_amdgcn_tensor_load_to_lds(g0, g1, z4, z4, 0);
#endif
}

// ---- main: WMMA scores, argmax over codes, gather + loss partials ----
__global__ __launch_bounds__(256)
void vq_main(const float* __restrict__ x,
             const float* __restrict__ cb,
             const unsigned short* __restrict__ cbhi,
             const unsigned short* __restrict__ cblo,
             const float* __restrict__ hn,
             float* __restrict__ out,
             float* __restrict__ partials) {
  __shared__ alignas(128) unsigned short s_hi[K_CODES * DIM]; // 128 KB bf16-hi codebook
  __shared__ float s_hn[K_CODES];                             // 4 KB
  __shared__ int   s_bestk[ROWS_PER_BLOCK];
  __shared__ float s_red[THREADS];

  const int tid = threadIdx.x;

  // Stage bf16-hi codebook into LDS via the Tensor Data Mover (wave 0 only).
  if (tid < 32) {
    tdm_stage_cbhi(cbhi, (unsigned)(uintptr_t)&s_hi[0]);
    __builtin_amdgcn_s_wait_tensorcnt(0);
  }
  for (int i = tid; i < K_CODES; i += THREADS) s_hn[i] = hn[i];
  __syncthreads();

  const int lane = tid & 31;
  const int wave = tid >> 5;
  const int half = lane >> 4;   // which 16-lane half
  const int nIdx = lane & 15;   // row (for A) / column (for B, C/D)
  const int rowBase = blockIdx.x * ROWS_PER_BLOCK + wave * ROWS_PER_WAVE;

  // Two 16-row A-fragment sets per wave (hi/lo bf16 split). Layout per lane
  // half h, element e: K = 32c + 16h + e (contiguous, matching B)
  V16U Ahi[2][2], Alo[2][2];
#pragma unroll
  for (int s = 0; s < 2; ++s) {
    const float* xr = x + (size_t)(rowBase + s * 16 + nIdx) * DIM;
#pragma unroll
    for (int c = 0; c < 2; ++c) {
      const int kb = 32 * c + half * 16;
#pragma unroll
      for (int e = 0; e < 8; ++e) {
        float v0 = xr[kb + e];
        float v1 = xr[kb + 8 + e];
        bf16 h0 = (bf16)v0, h1 = (bf16)v1;
        Ahi[s][c].v[e]     = h0;  Alo[s][c].v[e]     = (bf16)(v0 - (float)h0);
        Ahi[s][c].v[8 + e] = h1;  Alo[s][c].v[8 + e] = (bf16)(v1 - (float)h1);
      }
    }
  }

  float best[2][8];
  int   bestk[2][8];
#pragma unroll
  for (int s = 0; s < 2; ++s)
#pragma unroll
    for (int r = 0; r < 8; ++r) { best[s][r] = -3.4e38f; bestk[s][r] = 0; }

  // Sweep 64 tiles of 16 codes; score t = x.c - 0.5||c||^2 via 12 bf16 WMMAs
  // (B fragments amortized over 32 rows).
  for (int t = 0; t < K_CODES / 16; ++t) {
    const int code = t * 16 + nIdx;
    V16U Bh0, Bh1, Bl0, Bl1;
    {
      const uint4* ph0 = (const uint4*)&s_hi[code * DIM + half * 16];
      const uint4* ph1 = (const uint4*)&s_hi[code * DIM + 32 + half * 16];
      Bh0.q[0] = ph0[0]; Bh0.q[1] = ph0[1];
      Bh1.q[0] = ph1[0]; Bh1.q[1] = ph1[1];
      const uint4* pl0 = (const uint4*)&cblo[code * DIM + half * 16];
      const uint4* pl1 = (const uint4*)&cblo[code * DIM + 32 + half * 16];
      Bl0.q[0] = pl0[0]; Bl0.q[1] = pl0[1];
      Bl1.q[0] = pl1[0]; Bl1.q[1] = pl1[1];
    }
    const float h = s_hn[code];
    v8f acc0, acc1;
#pragma unroll
    for (int i = 0; i < 8; ++i) { acc0[i] = h; acc1[i] = h; }  // fold -0.5||c||^2
    acc0 = WMMA_BF16(Ahi[0][0].v, Bh0.v, acc0);
    acc1 = WMMA_BF16(Ahi[1][0].v, Bh0.v, acc1);
    acc0 = WMMA_BF16(Ahi[0][1].v, Bh1.v, acc0);
    acc1 = WMMA_BF16(Ahi[1][1].v, Bh1.v, acc1);
    acc0 = WMMA_BF16(Alo[0][0].v, Bh0.v, acc0);
    acc1 = WMMA_BF16(Alo[1][0].v, Bh0.v, acc1);
    acc0 = WMMA_BF16(Alo[0][1].v, Bh1.v, acc0);
    acc1 = WMMA_BF16(Alo[1][1].v, Bh1.v, acc1);
    acc0 = WMMA_BF16(Ahi[0][0].v, Bl0.v, acc0);
    acc1 = WMMA_BF16(Ahi[1][0].v, Bl0.v, acc1);
    acc0 = WMMA_BF16(Ahi[0][1].v, Bl1.v, acc0);
    acc1 = WMMA_BF16(Ahi[1][1].v, Bl1.v, acc1);
#pragma unroll
    for (int r = 0; r < 8; ++r) {
      if (acc0[r] > best[0][r]) { best[0][r] = acc0[r]; bestk[0][r] = code; }
      if (acc1[r] > best[1][r]) { best[1][r] = acc1[r]; bestk[1][r] = code; }
    }
  }

  // Cross-lane argmax within each 16-lane half (D layout: M = r + 8*half, N = nIdx).
#pragma unroll
  for (int s = 0; s < 2; ++s) {
#pragma unroll
    for (int r = 0; r < 8; ++r) {
      float v = best[s][r];
      int   k = bestk[s][r];
#pragma unroll
      for (int m = 8; m >= 1; m >>= 1) {
        float ov = __shfl_xor(v, m, 16);
        int   ok = __shfl_xor(k, m, 16);
        if (ov > v || (ov == v && ok < k)) { v = ov; k = ok; }  // first-index tie-break
      }
      if (nIdx == 0) s_bestk[wave * ROWS_PER_WAVE + s * 16 + half * 8 + r] = k;
    }
  }
  __syncthreads();

  // Gather exact f32 codebook rows, write quantized output, accumulate loss.
  float lsum = 0.f;
  const size_t base = (size_t)blockIdx.x * (ROWS_PER_BLOCK * DIM);
  for (int i = tid; i < ROWS_PER_BLOCK * DIM; i += THREADS) {
    const int row = i >> 6, d = i & 63;
    const int k = s_bestk[row];
    const float c = cb[k * DIM + d];
    const float xv = x[base + i];
    out[base + i] = c;
    const float df = c - xv;
    lsum += df * df;
  }
  s_red[tid] = lsum;
  __syncthreads();
#pragma unroll
  for (int s = THREADS / 2; s > 0; s >>= 1) {
    if (tid < s) s_red[tid] += s_red[tid + s];
    __syncthreads();
  }
  if (tid == 0) partials[blockIdx.x] = s_red[0];
}

// ---- deterministic final loss reduce: 1.25 * mean((q - x)^2) ----
__global__ __launch_bounds__(256)
void vq_reduce(const float* __restrict__ partials, float* __restrict__ out_scalar) {
  __shared__ float s_red[256];
  float s = 0.f;
  for (int i = threadIdx.x; i < MAIN_BLOCKS; i += 256) s += partials[i];
  s_red[threadIdx.x] = s;
  __syncthreads();
  for (int k = 128; k > 0; k >>= 1) {
    if (threadIdx.x < k) s_red[threadIdx.x] += s_red[threadIdx.x + k];
    __syncthreads();
  }
  if (threadIdx.x == 0)
    *out_scalar = 1.25f * s_red[0] / (float)(N_ROWS * DIM);
}

extern "C" void kernel_launch(void* const* d_in, const int* in_sizes, int n_in,
                              void* d_out, int out_size, void* d_ws, size_t ws_size,
                              hipStream_t stream) {
  const float* x  = (const float*)d_in[0];   // [65536, 64] f32
  const float* cb = (const float*)d_in[1];   // [1024, 64] f32
  char* ws = (char*)d_ws;
  unsigned short* cbhi = (unsigned short*)(ws);            // 128 KB
  unsigned short* cblo = (unsigned short*)(ws + 131072);   // 128 KB
  float* hn            = (float*)(ws + 262144);            // 4 KB
  float* partials      = (float*)(ws + 266240);            // 1 KB
  float* out = (float*)d_out;
  float* out_scalar = out + (size_t)N_ROWS * DIM;

  vq_prep_split<<<(K_CODES * DIM) / 256, 256, 0, stream>>>(cb, cbhi, cblo);
  vq_prep_norm<<<(K_CODES + 255) / 256, 256, 0, stream>>>(cb, hn);
  vq_main<<<MAIN_BLOCKS, THREADS, 0, stream>>>(x, cb, cbhi, cblo, hn, out, partials);
  vq_reduce<<<1, 256, 0, stream>>>(partials, out_scalar);
}
